// ConvLayer_40123584479572
// MI455X (gfx1250) — matrix operand order repaired and checked by reading, not compiled, not run on hardware
//
#include <hip/hip_runtime.h>

typedef __attribute__((ext_vector_type(16))) _Float16 v16h;
typedef __attribute__((ext_vector_type(8)))  _Float16 v8h;
typedef __attribute__((ext_vector_type(8)))  float    v8f;
typedef __attribute__((ext_vector_type(4)))  float    v4f;

#define NNODES   40000
#define FDIM     128
#define BDIM     64
#define NEDGES   640000
#define KDIM     320          // 2F + B
#define NDIM     256          // 2F
#define WT_STRIDE 328         // halves per W column (320 + 8 pad): stride 656B -> conflict-free ds_load_b128
#define LDS_BYTES (NDIM * WT_STRIDE * 2)   // 167,936 B dynamic LDS (W tile); stats live in static LDS

// ---- helpers -------------------------------------------------------------

__device__ __forceinline__ v8h cvt8(const float* __restrict__ p) {
  v4f x0 = *(const v4f*)(p);
  v4f x1 = *(const v4f*)(p + 4);
  v8h r;
  r[0] = (_Float16)x0[0]; r[1] = (_Float16)x0[1];
  r[2] = (_Float16)x0[2]; r[3] = (_Float16)x0[3];
  r[4] = (_Float16)x1[0]; r[5] = (_Float16)x1[1];
  r[6] = (_Float16)x1[2]; r[7] = (_Float16)x1[3];
  return r;
}

// ---- kernel 1: fused gather + f16 WMMA GEMM + column sum/sumsq -----------
// One wave owns a 16-edge row tile; iterates the 16 column tiles two at a
// time -> two independent WMMA accumulator chains in flight per wave.
// A fragments gathered straight from global (v_i | v_j | e_ij); W f16 in LDS.
// Column statistics accumulate via ds_add_f32 into static LDS, flushed once.

__global__ __launch_bounds__(256, 1) void gemm_stats_kernel(
    const float* __restrict__ atom, const float* __restrict__ edgef,
    const long long* __restrict__ eidx, const float* __restrict__ W,
    _Float16* __restrict__ y, float* __restrict__ msg_sum, float* __restrict__ msg_sq,
    int ntiles)
{
  extern __shared__ char smem[];
  _Float16* Wt = (_Float16*)smem;
  __shared__ float lsum[NDIM];
  __shared__ float lsq[NDIM];

  // zero LDS stats (blockDim.x == NDIM == 256)
  lsum[threadIdx.x] = 0.f;
  lsq[threadIdx.x]  = 0.f;

  // Stage W (320x256 f32, row-major) -> LDS, transposed to column-major f16 w/ pad.
  for (int i = threadIdx.x; i < KDIM * NDIM; i += 256) {
    int k = i >> 8;
    int n = i & 255;
    Wt[n * WT_STRIDE + k] = (_Float16)W[i];
  }
  __syncthreads();

  const int lane  = threadIdx.x & 31;
  const int gw    = blockIdx.x * 8 + (threadIdx.x >> 5);
  const int gws   = gridDim.x * 8;
  const int m     = lane & 15;
  const int kbase = (lane >> 4) * 16;   // A fragment: K run of 16 per lane half
  const int bkoff = (lane >> 4) * 16;   // B fragment: K run of 16 per lane half
  const int ncol  = lane & 15;

  for (int tile = gw; tile < ntiles; tile += gws) {   // uniform trip count: EXEC all-ones for WMMA
    const int e = tile * 16 + m;
    const long long s = eidx[2 * e];
    const long long d = eidx[2 * e + 1];
    const float* pvi = atom + s * FDIM;
    const float* pvj = atom + d * FDIM;
    const float* pef = edgef + (long long)e * BDIM;

    v16h a[10];
    #pragma unroll
    for (int kt = 0; kt < 10; ++kt) {
      const int kr = kt * 32;   // each 32-wide k-tile lies wholly in one of v_i | v_j | e_ij
      const float* base = (kr < FDIM)     ? (pvi + kr)
                        : (kr < 2 * FDIM) ? (pvj + (kr - FDIM))
                                          : (pef + (kr - 2 * FDIM));
      ((v8h*)&a[kt])[0] = cvt8(base + kbase);
      ((v8h*)&a[kt])[1] = cvt8(base + kbase + 8);
    }

    #pragma unroll 1          // rolled: exactly two accumulator chains live
    for (int nt = 0; nt < 16; nt += 2) {
      v8f acc0 = {};
      v8f acc1 = {};
      const _Float16* bp0 = Wt + (nt * 16 + ncol) * WT_STRIDE + bkoff;
      const _Float16* bp1 = bp0 + 16 * WT_STRIDE;
      #pragma unroll
      for (int kt = 0; kt < 10; ++kt) {
        v16h b0, b1;
        ((v8h*)&b0)[0] = *(const v8h*)(bp0 + kt * 32);
        ((v8h*)&b0)[1] = *(const v8h*)(bp0 + kt * 32 + 8);
        ((v8h*)&b1)[0] = *(const v8h*)(bp1 + kt * 32);
        ((v8h*)&b1)[1] = *(const v8h*)(bp1 + kt * 32 + 8);
        acc0 = __builtin_amdgcn_wmma_f32_16x16x32_f16(false, a[kt], false, b0,
                                                      (short)0, acc0, false, false);
        acc1 = __builtin_amdgcn_wmma_f32_16x16x32_f16(false, a[kt], false, b1,
                                                      (short)0, acc1, false, false);
      }
      // store y tiles (f16) + lane-local stat partials
      const long long rb = ((long long)tile * 16 + 8 * (lane >> 4)) * NDIM + nt * 16 + ncol;
      float s0 = 0.f, q0 = 0.f, s1 = 0.f, q1 = 0.f;
      #pragma unroll
      for (int r = 0; r < 8; ++r) {
        float v0 = acc0[r];
        float v1 = acc1[r];
        s0 += v0; q0 += v0 * v0;
        s1 += v1; q1 += v1 * v1;
        y[rb + (long long)r * NDIM]      = (_Float16)v0;
        y[rb + (long long)r * NDIM + 16] = (_Float16)v1;
      }
      // lanes L and L+16 share a column: 2-way serialized ds_add_f32, still cheap
      atomicAdd(&lsum[nt * 16 + ncol],      s0);
      atomicAdd(&lsq[nt * 16 + ncol],       q0);
      atomicAdd(&lsum[nt * 16 + 16 + ncol], s1);
      atomicAdd(&lsq[nt * 16 + 16 + ncol],  q1);
    }
  }

  __syncthreads();
  // one global atomic per column per workgroup
  atomicAdd(msg_sum + threadIdx.x, lsum[threadIdx.x]);
  atomicAdd(msg_sq  + threadIdx.x, lsq[threadIdx.x]);
}

// ---- kernel 2/5: fold sums into (scale, shift); linear bias cancels in BN.

__global__ void finalize_stats_kernel(const float* __restrict__ sum, const float* __restrict__ sq,
                                      const float* __restrict__ gamma, const float* __restrict__ beta,
                                      float* __restrict__ scale, float* __restrict__ shift,
                                      float invM, int C)
{
  int c = blockIdx.x * blockDim.x + threadIdx.x;
  if (c >= C) return;
  float mean = sum[c] * invM;
  float var  = sq[c] * invM - mean * mean;
  float sc   = gamma[c] * rsqrtf(var + 1e-5f);
  scale[c] = sc;
  shift[c] = beta[c] - mean * sc;
}

// ---- kernel 3: BN apply + sigmoid*softplus + segment_sum scatter ---------

__global__ __launch_bounds__(256) void gate_scatter_kernel(
    const _Float16* __restrict__ y, const long long* __restrict__ eidx,
    const float* __restrict__ scale, const float* __restrict__ shift,
    float* __restrict__ m_i)
{
  long long i = (long long)blockIdx.x * 256 + threadIdx.x;  // exactly E*128 threads
  int c = (int)(i & 127);
  long long e = i >> 7;
  float f = (float)y[e * NDIM + c]        * scale[c]        + shift[c];
  float s = (float)y[e * NDIM + FDIM + c] * scale[FDIM + c] + shift[FDIM + c];
  float sig = 1.0f / (1.0f + __expf(-f));
  float sp  = fmaxf(s, 0.f) + log1pf(__expf(-fabsf(s)));
  long long src = eidx[2 * e];
  atomicAdd(m_i + src * FDIM + c, sig * sp);   // no-return f32 atomic -> STOREcnt path
}

// ---- kernel 4: per-column stats of m_i over nodes ------------------------

__global__ __launch_bounds__(256) void agg_stats_kernel(const float* __restrict__ m_i,
                                                        float* __restrict__ asum,
                                                        float* __restrict__ asq)
{
  __shared__ float ls[256], lq[256];
  const long long total = (long long)NNODES * FDIM;
  float s = 0.f, q = 0.f;
  for (long long i = (long long)blockIdx.x * 256 + threadIdx.x; i < total;
       i += (long long)gridDim.x * 256) {       // stride multiple of 128 keeps column fixed
    float v = m_i[i]; s += v; q += v * v;
  }
  ls[threadIdx.x] = s; lq[threadIdx.x] = q;
  __syncthreads();
  if (threadIdx.x < FDIM) {
    atomicAdd(asum + threadIdx.x, ls[threadIdx.x] + ls[threadIdx.x + FDIM]);
    atomicAdd(asq  + threadIdx.x, lq[threadIdx.x] + lq[threadIdx.x + FDIM]);
  }
}

// ---- kernel 6: out = softplus(atom + BN(m_i)) ----------------------------

__global__ __launch_bounds__(256) void output_kernel(const float* __restrict__ atom,
                                                     const float* __restrict__ m_i,
                                                     const float* __restrict__ scale,
                                                     const float* __restrict__ shift,
                                                     float* __restrict__ out)
{
  long long i = (long long)blockIdx.x * 256 + threadIdx.x;  // exactly N*128
  int c = (int)(i & 127);
  float a = atom[i] + m_i[i] * scale[c] + shift[c];
  out[i] = fmaxf(a, 0.f) + log1pf(__expf(-fabsf(a)));
}

// ---- launcher ------------------------------------------------------------

extern "C" void kernel_launch(void* const* d_in, const int* in_sizes, int n_in,
                              void* d_out, int out_size, void* d_ws, size_t ws_size,
                              hipStream_t stream)
{
  (void)in_sizes; (void)n_in; (void)out_size; (void)ws_size;
  const float* atom      = (const float*)d_in[0];
  const float* edgef     = (const float*)d_in[1];
  const long long* eidx  = (const long long*)d_in[2];
  const float* W         = (const float*)d_in[3];
  // d_in[4] = linear bias b : mathematically cancels inside batchnorm, unused.
  const float* g_msg     = (const float*)d_in[5];
  const float* b_msg     = (const float*)d_in[6];
  const float* g_agg     = (const float*)d_in[7];
  const float* b_agg     = (const float*)d_in[8];
  float* out = (float*)d_out;

  char* ws = (char*)d_ws;
  _Float16* y = (_Float16*)ws;                               // E*256 f16  (~328 MB)
  size_t off = (size_t)NEDGES * NDIM * sizeof(_Float16);
  float* m_i = (float*)(ws + off);                           // N*128 f32  (~20.5 MB)
  off += (size_t)NNODES * FDIM * sizeof(float);
  float* stats = (float*)(ws + off);                         // 1536 f32
  float* msg_sum   = stats;
  float* msg_sq    = stats + 256;
  float* agg_sum   = stats + 512;
  float* agg_sq    = stats + 640;
  float* msg_scale = stats + 768;
  float* msg_shift = stats + 1024;
  float* agg_scale = stats + 1280;
  float* agg_shift = stats + 1408;

  hipMemsetAsync(m_i, 0, (size_t)NNODES * FDIM * sizeof(float), stream);
  hipMemsetAsync(stats, 0, 768 * sizeof(float), stream);

  gemm_stats_kernel<<<512, 256, LDS_BYTES, stream>>>(atom, edgef, eidx, W, y,
                                                     msg_sum, msg_sq, NEDGES / 16);
  finalize_stats_kernel<<<1, 256, 0, stream>>>(msg_sum, msg_sq, g_msg, b_msg,
                                               msg_scale, msg_shift, 1.0f / NEDGES, NDIM);
  gate_scatter_kernel<<<(NEDGES * FDIM) / 256, 256, 0, stream>>>(y, eidx, msg_scale,
                                                                 msg_shift, m_i);
  agg_stats_kernel<<<512, 256, 0, stream>>>(m_i, agg_sum, agg_sq);
  finalize_stats_kernel<<<1, 128, 0, stream>>>(agg_sum, agg_sq, g_agg, b_agg,
                                               agg_scale, agg_shift, 1.0f / NNODES, FDIM);
  output_kernel<<<(NNODES * FDIM) / 256, 256, 0, stream>>>(atom, m_i, agg_scale,
                                                           agg_shift, out);
}
